// StructureAwareTensorSketch_21818433864274
// MI455X (gfx1250) — matrix operand order repaired and checked by reading, not compiled, not run on hardware
//
#include <hip/hip_runtime.h>
#include <hip/hip_bf16.h>

// ---------------------------------------------------------------------------
// StructureAwareTensorSketch for MI455X (gfx1250, wave32)
// Prep   : swizzle Wh/Ws1 into WMMA bf16 fragment layout (zero-padded K=32)
// Phase 1: WMMA bf16 (16x16x32) featurizer -> hash_probs (bf16) + sign (f32)
//          native v_tanh_f32; fragment loads are straight b128 vector loads
// Phase 2: single-WGP persistent scan; per-step prob tile streamed by the
//          Tensor Data Mover (tensor_load_to_lds + s_wait_tensorcnt),
//          double-buffered one step ahead.
// ---------------------------------------------------------------------------

typedef __attribute__((ext_vector_type(16))) __bf16 v16bf;
typedef __attribute__((ext_vector_type(8)))  float  v8f;

#define ALPHABET 4
#define CTX 2
#define WIN 5
#define CIN 20          // ALPHABET*WIN
#define DD  128
#define LL  3

#define HAVE_TDM __has_builtin(__builtin_amdgcn_tensor_load_to_lds)

#if HAVE_TDM
typedef __attribute__((ext_vector_type(4))) unsigned int tdm_g0_t;
typedef __attribute__((ext_vector_type(8))) int          tdm_g1_t;
typedef __attribute__((ext_vector_type(4))) int          tdm_g2_t;
typedef __attribute__((ext_vector_type(8))) int          tdm_g4_t;
#endif

static __device__ __forceinline__ float fast_tanh(float x) {
#if __has_builtin(__builtin_amdgcn_tanhf)
    return __builtin_amdgcn_tanhf(x);           // v_tanh_f32
#elif __has_builtin(__builtin_amdgcn_tanh_f32)
    return __builtin_amdgcn_tanh_f32(x);
#else
    return tanhf(x);
#endif
}

static __device__ __forceinline__ unsigned short f2bf(float f) {
    unsigned int u = __float_as_uint(f);
    unsigned int r = (u + 0x7fffu + ((u >> 16) & 1u)) >> 16;   // RNE
    return (unsigned short)r;
}

// ---------------------------------------------------------------------------
// Prep: build per-lane WMMA B-fragments in memory.
//   wh_swz : [L][8 n-tiles][32 lanes][16 elems] bf16  (24 KB)
//   ws1_swz: [L][32 lanes][16 elems]            bf16  ( 3 KB)
// Fragment element e of lane l holds K = (e<8 ? 0 : 16) + (l<16?0:8) + (e&7), N = ntile*16 + (l&15),
// zero for K >= CIN.
// ---------------------------------------------------------------------------
__global__ __launch_bounds__(64)
void sats_prep(const float* __restrict__ Wh, const float* __restrict__ Ws1,
               unsigned short* __restrict__ wh_swz,
               unsigned short* __restrict__ ws1_swz)
{
    int t = blockIdx.x * 64 + threadIdx.x;
    if (t < LL * 8 * 32) {
        int lane = t & 31, n = (t >> 5) & 7, p = t / (8 * 32);
        int half = lane >> 4, mrow = lane & 15;
        int N = n * 16 + mrow;
#pragma unroll
        for (int e = 0; e < 16; ++e) {
            int K = (e < 8) ? half * 8 + e : 16 + half * 8 + (e - 8);
            float x = (K < CIN) ? Wh[((size_t)p * CIN + K) * DD + N] : 0.f;
            wh_swz[(size_t)t * 16 + e] = f2bf(x);
        }
    }
    if (t < LL * 32) {
        int lane = t & 31, p = t >> 5;
        int half = lane >> 4, mrow = lane & 15;
#pragma unroll
        for (int e = 0; e < 16; ++e) {
            int K = (e < 8) ? half * 8 + e : 16 + half * 8 + (e - 8);
            float x = (K < CIN) ? Ws1[((size_t)p * CIN + K) * 16 + mrow] : 0.f;
            ws1_swz[(size_t)t * 16 + e] = f2bf(x);
        }
    }
}

// ---------------------------------------------------------------------------
// Phase 1: one wave (32 threads) handles one 16-row tile for one layer p.
// ---------------------------------------------------------------------------
__global__ __launch_bounds__(32)
void sats_phase1(const int* __restrict__ seq,
                 const unsigned short* __restrict__ wh_swz,
                 const float* __restrict__ bh,
                 const unsigned short* __restrict__ ws1_swz,
                 const float* __restrict__ bs1,
                 const float* __restrict__ Ws2, const float* __restrict__ bs2,
                 unsigned short* __restrict__ hpws,  // (L,S,D) bf16 bits
                 float* __restrict__ sws,            // (L,S)   f32
                 int S)
{
    const int lane = threadIdx.x;        // 0..31
    const int half = lane >> 4;          // 0 or 1
    const int mrow = lane & 15;
    const int tile = blockIdx.x;
    const int p    = blockIdx.y;
    const int srow = tile * 16 + mrow;   // A-matrix row for this lane

    // ---- Window symbols: 5 clamped (unconditional) loads, one wait --------
    int sv[WIN];
#pragma unroll
    for (int w = 0; w < WIN; ++w) {
        int pos  = srow + w - CTX;
        int posc = pos < 0 ? 0 : (pos >= S ? S - 1 : pos);
        int val  = seq[posc];
        sv[w] = (pos >= 0 && pos < S) ? val : -1;
    }

    // ---- Build A fragment (16x32 bf16), branch-free -----------------------
    // After unrolling, per-element alpha/window/pad selectors are constants;
    // only the half-select is runtime.
    v16bf afrag;
#pragma unroll
    for (int e = 0; e < 16; ++e) {
        const int v  = e >> 1;
        const int a2 = ((v & 1) << 1) | (e & 1);             // K & 3
        const int w0 = ((v >> 2) << 2) | ((v & 3) >> 1);     // window, half==0
        const int w1 = w0 + 2;                               // window, half==1
        float x0 = (w0 < WIN) ? ((sv[w0 < WIN ? w0 : 0] == a2) ? 1.f : 0.f) : 0.f;
        float x1 = (w1 < WIN) ? ((sv[w1 < WIN ? w1 : 0] == a2) ? 1.f : 0.f) : 0.f;
        afrag[e] = (__bf16)(half ? x1 : x0);
    }

    const v16bf* whf  = (const v16bf*)wh_swz  + (size_t)p * 8 * 32;
    const v16bf* ws1f = (const v16bf*)ws1_swz + (size_t)p * 32;
    const float* bhp  = bh + (size_t)p * DD;

    // ---- 8 WMMAs across N=128; keep tanh(logit+bias) in registers ---------
    float tval[8][8];            // [n-tile][C row r]
#pragma unroll
    for (int n = 0; n < 8; ++n) {
        v16bf bfrag = whf[(size_t)n * 32 + lane];            // 2x b128 load
        v8f c = {0.f,0.f,0.f,0.f,0.f,0.f,0.f,0.f};
        c = __builtin_amdgcn_wmma_f32_16x16x32_bf16(
                false, afrag, false, bfrag, (short)0, c, false, false);
        int N = n * 16 + mrow;
        float bias = bhp[N];
#pragma unroll
        for (int r = 0; r < 8; ++r) tval[n][r] = fast_tanh(c[r] + bias);
    }

    // ---- Row softmax: each row M lives in one 16-lane half ----------------
#pragma unroll
    for (int r = 0; r < 8; ++r) {
        float mx = tval[0][r];
#pragma unroll
        for (int n = 1; n < 8; ++n) mx = fmaxf(mx, tval[n][r]);
#pragma unroll
        for (int m = 8; m >= 1; m >>= 1) mx = fmaxf(mx, __shfl_xor(mx, m, 32));
        float ex[8];
        float lsum = 0.f;
#pragma unroll
        for (int n = 0; n < 8; ++n) { ex[n] = __expf(tval[n][r] - mx); lsum += ex[n]; }
#pragma unroll
        for (int m = 8; m >= 1; m >>= 1) lsum += __shfl_xor(lsum, m, 32);
        float inv = 1.f / lsum;
        int row = half * 8 + r;
        size_t base = ((size_t)p * S + (size_t)(tile * 16 + row)) * DD;
#pragma unroll
        for (int n = 0; n < 8; ++n)
            hpws[base + n * 16 + mrow] = f2bf(ex[n] * inv);
    }

    // ---- Sign head: one more WMMA (N=16) ----------------------------------
    v16bf b2 = ws1f[lane];
    v8f c2 = {0.f,0.f,0.f,0.f,0.f,0.f,0.f,0.f};
    c2 = __builtin_amdgcn_wmma_f32_16x16x32_bf16(
            false, afrag, false, b2, (short)0, c2, false, false);
    float w2  = Ws2[(size_t)p * 16 + mrow];
    float b1v = bs1[(size_t)p * 16 + mrow];
    float b2v = bs2[p];
#pragma unroll
    for (int r = 0; r < 8; ++r) {
        float t = fmaxf(c2[r] + b1v, 0.f) * w2;
#pragma unroll
        for (int m = 8; m >= 1; m >>= 1) t += __shfl_xor(t, m, 32);
        if (mrow == 0) {
            int row = half * 8 + r;
            float sg = 1.f / (1.f + __expf(-fast_tanh(t + b2v)));
            sws[(size_t)p * S + (size_t)(tile * 16 + row)] = sg;
        }
    }
}

// ---------------------------------------------------------------------------
// Phase 2: serial scan, one workgroup (512 threads, 16 waves on one WGP).
// ---------------------------------------------------------------------------
__global__ __launch_bounds__(512)
void sats_phase2(const unsigned short* __restrict__ hpws,
                 const float* __restrict__ sws,
                 float* __restrict__ out, int S)
{
    __shared__ float Tq[2][DD][4];      // [pm][j][layer 0..3]
    __shared__ float hp4[DD][4];        // [s][layer 0..2], [3] stays 0
    __shared__ float part[4][3][DD];    // [group][layer][j]
    __shared__ float spv[4];
#if HAVE_TDM
    __shared__ unsigned short hpraw[2][LL * DD];   // TDM double buffer
#endif

    const int tid = threadIdx.x;
    for (int k = tid; k < 2 * DD * 4; k += 512) (&Tq[0][0][0])[k] = 0.f;
    for (int k = tid; k < DD * 4;     k += 512) (&hp4[0][0])[k]   = 0.f;
    if (tid == 0) Tq[0][0][0] = 1.f;    // Tp[layer0][j=0] = 1
    __syncthreads();

    const int grp = tid >> 7;           // 0..3
    const int j   = tid & 127;
    const int pm  = grp & 1;            // 0=Tp, 1=Tm
    const int s0  = (grp >> 1) * 64;    // K-split half

#if HAVE_TDM
    // ---- Tensor Data Mover descriptor (2D tile: 128 x 3, 2-byte elems) ----
    const unsigned long long row_stride = (unsigned long long)S * DD; // elems
    tdm_g1_t g1;
    g1[0] = (int)(1u << 16);                 // workgroup_mask=0, data_size=1 (2B)
    g1[1] = (int)((unsigned)DD << 16);       // tensor_dim0 = 128 (bits 63:48)
    g1[2] = (int)(3u << 16);                 // tensor_dim1 = 3   (bits 111:80 lo)
    g1[3] = (int)((unsigned)DD << 16);       // tile_dim0 = 128   (bits 127:112)
    g1[4] = 3;                               // tile_dim1 = 3, tile_dim2 = 0
    g1[5] = (int)(unsigned)(row_stride & 0xffffffffu);   // dim0_stride lo
    g1[6] = (int)(unsigned)(row_stride >> 32);           // dim0_stride hi
    g1[7] = 0;
    const tdm_g2_t gz = {0, 0, 0, 0};
    const unsigned lds_base[2] = {
        (unsigned)(size_t)&hpraw[0][0],
        (unsigned)(size_t)&hpraw[1][0]
    };
    const unsigned long long gbase = (unsigned long long)(size_t)hpws;

    auto tdm_issue = [&](int step, int buf) {
        unsigned long long ga = gbase + (unsigned long long)step * (DD * 2u);
        tdm_g0_t g0;
        g0[0] = 1u;                                        // count=1, user desc
        g0[1] = lds_base[buf];                             // lds_addr
        g0[2] = (unsigned)(ga & 0xffffffffu);              // global_addr lo
        g0[3] = (unsigned)((ga >> 32) & 0x01ffffffu) | (2u << 30); // hi + type=2
#if __clang_major__ >= 23
        tdm_g4_t gz8 = {0,0,0,0,0,0,0,0};
        __builtin_amdgcn_tensor_load_to_lds(g0, g1, gz, gz, gz8, 0);
#else
        __builtin_amdgcn_tensor_load_to_lds(g0, g1, gz, gz, 0);
#endif
    };

    if (tid < 32) tdm_issue(0, 0);      // prologue: fetch step 0
#endif

    for (int i = 0; i < S; ++i) {
#if HAVE_TDM
        if (tid < 32) {                 // wave 0: prefetch next, wait current
            if (i + 1 < S) {
                tdm_issue(i + 1, (i + 1) & 1);
                __builtin_amdgcn_s_wait_tensorcnt((short)1);
            } else {
                __builtin_amdgcn_s_wait_tensorcnt((short)0);
            }
        }
        __syncthreads();                // buf[i&1] ready; prev update visible
        if (tid < 384) {
            unsigned int raw = hpraw[i & 1][tid];
            hp4[tid & 127][tid >> 7] = __uint_as_float(raw << 16);
        } else if (tid < 387) {
            spv[tid - 384] = sws[(size_t)(tid - 384) * S + i];
        }
#else
        __syncthreads();                // prev update visible
        if (tid < 384) {
            int pl = tid >> 7, n = tid & 127;
            unsigned int raw = hpws[((size_t)pl * S + i) * DD + n];
            hp4[n][pl] = __uint_as_float(raw << 16);
        } else if (tid < 387) {
            spv[tid - 384] = sws[(size_t)(tid - 384) * S + i];
        } else if (tid >= 504 && i + 8 < S) {
            int q = tid - 504;
            int pl = q >> 1;
            if (pl < 3)
                __builtin_prefetch(hpws + ((size_t)pl * S + (size_t)(i + 8)) * DD
                                        + (q & 1) * 64, 0, 0);
        }
#endif
        __syncthreads();

        // ---- 6 circular convolutions, 2-way K split -----------------------
        const float* T = &Tq[pm][0][0];
        float a0 = 0.f, a1 = 0.f, a2 = 0.f;
#pragma unroll 4
        for (int s = s0; s < s0 + 64; ++s) {
            int idx = (j - s) & 127;
            float4 tv = *(const float4*)(T + idx * 4);       // ds_load_b128
            float4 hv = *(const float4*)(&hp4[s][0]);        // broadcast
            a0 += tv.x * hv.x;
            a1 += tv.y * hv.y;
            a2 += tv.z * hv.z;
        }
        part[grp][0][j] = a0;
        part[grp][1][j] = a1;
        part[grp][2][j] = a2;
        __syncthreads();

        // ---- affine update of layers 1..3 ---------------------------------
        if (tid < 384) {
            int pu = (tid >> 7) + 1;           // 1..3
            int jj = tid & 127;
            if (i + 1 >= pu) {
                int pidx = pu - 1;
                float shp = part[0][pidx][jj] + part[2][pidx][jj];
                float shm = part[1][pidx][jj] + part[3][pidx][jj];
                float z   = (float)pu / (float)(i + 1);
                float sg  = spv[pidx];
                float op = Tq[0][jj][pu], om = Tq[1][jj][pu];
                Tq[0][jj][pu] = (1.f - z) * op + z * (sg * shp + (1.f - sg) * shm);
                Tq[1][jj][pu] = (1.f - z) * om + z * (sg * shm + (1.f - sg) * shp);
            }
        }
    }
    __syncthreads();
    if (tid < DD) out[tid] = Tq[0][tid][3] - Tq[1][tid][3];
}

// ---------------------------------------------------------------------------
extern "C" void kernel_launch(void* const* d_in, const int* in_sizes, int n_in,
                              void* d_out, int out_size, void* d_ws, size_t ws_size,
                              hipStream_t stream)
{
    const int*   seq = (const int*)  d_in[0];
    const float* Wh  = (const float*)d_in[1];
    const float* bh  = (const float*)d_in[2];
    const float* Ws1 = (const float*)d_in[3];
    const float* bs1 = (const float*)d_in[4];
    const float* Ws2 = (const float*)d_in[5];
    const float* bs2 = (const float*)d_in[6];
    const int S = in_sizes[0];

    // workspace layout: hp (bf16) | sign (f32) | swizzled weights (bf16)
    char* ws = (char*)d_ws;
    unsigned short* hpws = (unsigned short*)ws;
    size_t off = ((size_t)LL * S * DD * sizeof(unsigned short) + 255) & ~(size_t)255;
    float* sws = (float*)(ws + off);
    off += ((size_t)LL * S * sizeof(float) + 255) & ~(size_t)255;
    unsigned short* wh_swz  = (unsigned short*)(ws + off);
    off += (size_t)LL * 8 * 32 * 16 * sizeof(unsigned short);
    unsigned short* ws1_swz = (unsigned short*)(ws + off);

    sats_prep<<<(LL * 8 * 32 + 63) / 64, 64, 0, stream>>>(Wh, Ws1, wh_swz, ws1_swz);
    dim3 g1(S / 16, LL);
    sats_phase1<<<g1, 32, 0, stream>>>(seq, wh_swz, bh, ws1_swz, bs1, Ws2, bs2,
                                       hpws, sws, S);
    sats_phase2<<<1, 512, 0, stream>>>(hpws, sws, (float*)d_out, S);
}
